// GraphSage_60756607369753
// MI455X (gfx1250) — matrix operand order repaired and checked by reading, not compiled, not run on hardware
//
#include <hip/hip_runtime.h>

typedef __attribute__((ext_vector_type(16))) _Float16 v16h;
typedef __attribute__((ext_vector_type(8)))  _Float16 v8h;
typedef __attribute__((ext_vector_type(4)))  _Float16 v4h;
typedef __attribute__((ext_vector_type(2)))  _Float16 v2h;
typedef __attribute__((ext_vector_type(8)))  float    v8f;
typedef __attribute__((ext_vector_type(4)))  float    v4f;

#define DIM  64
#define DEG  32
#define TOPK 16
#define WPB  4          // waves (nodes) per block

// LDS row pitches in halfs: byte pitch must be 16B-divisible for b128 frag
// reads, and staggered mod 64 banks. PA: 144 B/row, PT: 80 B/row.
#define PA 72
#define PT 40

struct alignas(16) WaveLDS {
    _Float16 tileA[DEG * PA];   // neighbors, row-major [32][64] f16 (padded)
    _Float16 tileT[DIM * PT];   // neighbors, transposed [64][32] f16 (padded)
    _Float16 uc[DIM];           // center embedding, f16
    _Float16 wts[DEG];          // top-k weights: 0 or 1/16 (exact in f16)
    float    dots[DEG];
    float    sq[DEG];
    float    sims[DEG];
};

union FragH { v16h v; v8h h[2]; };

__global__ __launch_bounds__(WPB * 32) void graphsage_topk_kernel(
    const int* __restrict__ nodes, const int* __restrict__ neigh,
    const float* __restrict__ vis, const float* __restrict__ txt,
    float* __restrict__ out, int n_nodes)
{
    __shared__ WaveLDS lds[WPB];
    const int wave = threadIdx.x >> 5;
    const int lane = threadIdx.x & 31;
    int node = blockIdx.x * WPB + wave;
    if (node >= n_nodes) node = n_nodes - 1;   // never taken for 50000/4; safe dup otherwise
    WaveLDS& W = lds[wave];

    const size_t NTOT = (size_t)n_nodes * DIM;
    const int uidx = nodes[node];
    const int nidx = neigh[(size_t)node * DEG + lane];
    const int m16  = lane & 15;
    const int hi   = lane >> 4;                // which half of the wave

#pragma unroll
    for (int m = 0; m < 2; ++m) {              // 0 = visual, 1 = text
        const float* __restrict__ tab = (m == 0) ? vis : txt;

        // ---- stage center (f32 -> output, f16 -> LDS broadcast array) ----
        const float2 cu = *(const float2*)(tab + (size_t)uidx * DIM + lane * 2);
        float* outC = out + (size_t)m * NTOT + (size_t)node * DIM;
        outC[lane * 2 + 0] = cu.x;
        outC[lane * 2 + 1] = cu.y;
        v2h ch; ch[0] = (_Float16)cu.x; ch[1] = (_Float16)cu.y;
        *(v2h*)&W.uc[lane * 2] = ch;

        // ---- stage this lane's neighbor row; f32 sum-of-squares on the fly ----
        const float* __restrict__ src = tab + (size_t)nidx * DIM;
        if (m == 0) __builtin_prefetch(txt + (size_t)nidx * DIM, 0, 1); // warm text row
        float sqa = 0.0f;
#pragma unroll
        for (int j = 0; j < 16; ++j) {
            const v4f v = *(const v4f*)(src + j * 4);      // global_load_b128 gather
            sqa = fmaf(v[0], v[0], sqa);
            sqa = fmaf(v[1], v[1], sqa);
            sqa = fmaf(v[2], v[2], sqa);
            sqa = fmaf(v[3], v[3], sqa);
            v4h hv; hv[0] = (_Float16)v[0]; hv[1] = (_Float16)v[1];
                    hv[2] = (_Float16)v[2]; hv[3] = (_Float16)v[3];
            *(v4h*)&W.tileA[lane * PA + j * 4] = hv;       // row-major tile
            W.tileT[(j * 4 + 0) * PT + lane] = hv[0];      // transposed tile
            W.tileT[(j * 4 + 1) * PT + lane] = hv[1];
            W.tileT[(j * 4 + 2) * PT + lane] = hv[2];
            W.tileT[(j * 4 + 3) * PT + lane] = hv[3];
        }
        W.sq[lane] = sqa;
        __syncthreads();

        // ---- sims: D = NB(16x32) x U_bcast(32x16), v_wmma_f32_16x16x32_f16 ----
        // A layout (ISA 7.12.2): lane<16 row m16 holds K = kbase+{0..7,16..23};
        // lane>=16 holds K = kbase+{8..15,24..31}.  B: same K order as A;
        // all 16 N columns = center (broadcast).
        v8f acc0 = {}; v8f acc1 = {};
#pragma unroll
        for (int c = 0; c < 2; ++c) {
            FragH B;
            const int kb = c * 32 + hi * 8;
            B.h[0] = *(const v8h*)&W.uc[kb];
            B.h[1] = *(const v8h*)&W.uc[kb + 16];
            const int ko = c * 32 + hi * 8;
            FragH A0, A1;
            A0.h[0] = *(const v8h*)&W.tileA[m16 * PA + ko];
            A0.h[1] = *(const v8h*)&W.tileA[m16 * PA + ko + 16];
            A1.h[0] = *(const v8h*)&W.tileA[(16 + m16) * PA + ko];
            A1.h[1] = *(const v8h*)&W.tileA[(16 + m16) * PA + ko + 16];
            acc0 = __builtin_amdgcn_wmma_f32_16x16x32_f16(false, A0.v, false, B.v,
                                                          (short)0, acc0, false, false);
            acc1 = __builtin_amdgcn_wmma_f32_16x16x32_f16(false, A1.v, false, B.v,
                                                          (short)0, acc1, false, false);
        }
        // D layout: VGPR r, lane l -> (M = r + 8*(l>=16), N = l%16); col 0 suffices
        if (m16 == 0) {
#pragma unroll
            for (int r = 0; r < 8; ++r) {
                W.dots[hi * 8 + r]      = acc0[r];   // neighbors  0..15
                W.dots[16 + hi * 8 + r] = acc1[r];   // neighbors 16..31
            }
        }
        __syncthreads();

        // ---- top-16 by rank; dropping (||u||+eps) keeps ordering identical ----
        const float mysim = W.dots[lane] / (sqrtf(W.sq[lane]) + 1e-8f);
        W.sims[lane] = mysim;
        __syncthreads();
        int rank = 0;
#pragma unroll
        for (int j = 0; j < DEG; ++j) {
            const float s = W.sims[j];
            rank += (s > mysim || (s == mysim && j < lane)) ? 1 : 0; // stable, lax.top_k order
        }
        W.wts[lane] = (rank < TOPK) ? (_Float16)0.0625f : (_Float16)0.0f;
        __syncthreads();

        // ---- aggregation: agg(1x64) = w(1x32) x NB(32x64); w broadcast to all A rows ----
        FragH Aw;
        const int kw = hi * 16;
        Aw.h[0] = *(const v8h*)&W.wts[kw];
        Aw.h[1] = *(const v8h*)&W.wts[kw + 8];
        float* outA = out + (size_t)(2 + m) * NTOT + (size_t)node * DIM;
#pragma unroll
        for (int nc = 0; nc < 4; ++nc) {
            FragH Bn;                                   // B[k][n] = nb_k[16*nc + n]
            const int row = nc * 16 + m16;
            const int k0  = hi * 16;
            Bn.h[0] = *(const v8h*)&W.tileT[row * PT + k0];
            Bn.h[1] = *(const v8h*)&W.tileT[row * PT + k0 + 8];
            v8f cacc = {};
            cacc = __builtin_amdgcn_wmma_f32_16x16x32_f16(false, Aw.v, false, Bn.v,
                                                          (short)0, cacc, false, false);
            if (lane < 16) outA[nc * 16 + lane] = cacc[0];  // every D row equals agg
        }
        __syncthreads();   // LDS reused by next modality
    }
}

extern "C" void kernel_launch(void* const* d_in, const int* in_sizes, int n_in,
                              void* d_out, int out_size, void* d_ws, size_t ws_size,
                              hipStream_t stream) {
    const int*   nodes = (const int*)d_in[0];
    const int*   neigh = (const int*)d_in[1];
    const float* vis   = (const float*)d_in[2];
    const float* txt   = (const float*)d_in[3];
    float*       out   = (float*)d_out;
    const int n_nodes  = in_sizes[0];
    const int blocks   = (n_nodes + WPB - 1) / WPB;     // 50000/4 = 12500 exact
    hipLaunchKernelGGL(graphsage_topk_kernel, dim3(blocks), dim3(WPB * 32), 0, stream,
                       nodes, neigh, vis, txt, out, n_nodes);
}
